// MultiHeadSelector_30210799960556
// MI455X (gfx1250) — compile-verified, matrix operand order checked by reading, not run on hardware
//
#include <hip/hip_runtime.h>
#include <hip/hip_bf16.h>
#include <math.h>

// Problem constants (from reference setup_inputs)
#define BATCH   8
#define CHAN    12
#define SPA     784          // S = 28*28
#define HH      28
#define NROW    785          // S+1
#define HID     768
#define HID1    512
#define KTOP    84
#define LEAKY_S 0.2f
#define SCALE_S 0.7f
#define INV_PI  0.31830988618379067f

typedef __attribute__((ext_vector_type(16))) _Float16 v16h;
typedef __attribute__((ext_vector_type(8)))  _Float16 v8h;
typedef __attribute__((ext_vector_type(8)))  float    v8f;

// ---------------- K0: zero the atomic count buffer ----------------
__global__ void k_zero(float* __restrict__ p, int n) {
    int i = blockIdx.x * blockDim.x + threadIdx.x;
    if (i < n) p[i] = 0.0f;
}

// ---------------- K1: per-(b,c) top-84 of score; new_score + count scatter ----
__global__ void k_score_topk(const float* __restrict__ x,
                             float* __restrict__ new_score,
                             float* __restrict__ count) {
    const int bc  = blockIdx.x;            // 0..95
    const int b   = bc / CHAN;
    const int tid = threadIdx.x;           // 256 threads
    __shared__ float s_orig[SPA];
    __shared__ float s_work[SPA];
    __shared__ int   s_sel[SPA];
    __shared__ float rv[256];
    __shared__ int   ri[256];

    const float* row = x + ((size_t)bc * NROW + 0) * NROW + 1;   // x[b,c,0,1:]
    for (int n = tid; n < SPA; n += 256) {
        float v = row[n];
        s_orig[n] = v; s_work[n] = v; s_sel[n] = 0;
    }
    __syncthreads();

    for (int it = 0; it < KTOP; ++it) {
        float bv = -INFINITY; int bi = 0x7fffffff;
        for (int n = tid; n < SPA; n += 256) {
            float v = s_work[n];
            if (v > bv || (v == bv && n < bi)) { bv = v; bi = n; }
        }
        rv[tid] = bv; ri[tid] = bi;
        __syncthreads();
        for (int off = 128; off > 0; off >>= 1) {
            if (tid < off) {
                float ov = rv[tid + off]; int oi = ri[tid + off];
                if (ov > rv[tid] || (ov == rv[tid] && oi < ri[tid])) { rv[tid] = ov; ri[tid] = oi; }
            }
            __syncthreads();
        }
        if (tid == 0) { int w = ri[0]; s_sel[w] = 1; s_work[w] = -INFINITY; }
        __syncthreads();
    }

    for (int n = tid; n < SPA; n += 256) {
        float v = s_orig[n];
        new_score[(size_t)bc * SPA + n] = s_sel[n] ? v : SCALE_S * v;
        if (s_sel[n]) atomicAdd(&count[b * SPA + n], 1.0f);   // exact: small ints
    }
}

// ---------------- K2: per-batch mask/thr/argmax/geometry + U vector -----------
__global__ void k_batch_reduce(const float* __restrict__ new_score,
                               const float* __restrict__ w1,     // (2,512)
                               float* __restrict__ U,            // (8,512)
                               float* __restrict__ panchor) {    // (8)
    const int b = blockIdx.x, tid = threadIdx.x;   // 8 blocks x 256
    __shared__ float pw [SPA];
    __shared__ float msk[SPA];
    __shared__ float red [256];
    __shared__ float red2[256];
    __shared__ int   redi[256];

    for (int n = tid; n < SPA; n += 256) {
        float m = 0.f;
        for (int c = 0; c < CHAN; ++c)
            m += new_score[((size_t)(b * CHAN + c)) * SPA + n];
        msk[n] = m; pw[n] = m * (1.0f / CHAN);
    }
    __syncthreads();

    // threshold = mean over n of msk
    float part = 0.f;
    for (int n = tid; n < SPA; n += 256) part += msk[n];
    red[tid] = part; __syncthreads();
    for (int off = 128; off > 0; off >>= 1) { if (tid < off) red[tid] += red[tid + off]; __syncthreads(); }
    const float thr = red[0] * (1.0f / SPA);
    __syncthreads();

    // argmax of pw * (msk > thr), first-occurrence tie-break
    float bv = -INFINITY; int bi = 0x7fffffff;
    for (int n = tid; n < SPA; n += 256) {
        float v = (msk[n] > thr) ? pw[n] : 0.0f;
        if (v > bv || (v == bv && n < bi)) { bv = v; bi = n; }
    }
    red[tid] = bv; redi[tid] = bi; __syncthreads();
    for (int off = 128; off > 0; off >>= 1) {
        if (tid < off) {
            float ov = red[tid + off]; int oi = redi[tid + off];
            if (ov > red[tid] || (ov == red[tid] && oi < redi[tid])) { red[tid] = ov; redi[tid] = oi; }
        }
        __syncthreads();
    }
    const int max_idx = redi[0];
    __syncthreads();

    const int ai = max_idx / HH, aj = max_idx % HH;
    // weighted sums of dist & angle features
    float sa = 0.f, sb = 0.f;
    for (int n = tid; n < SPA; n += 256) {
        float rI = (float)(n / HH - ai) * (1.0f / HH);
        float rJ = (float)(n % HH - aj) * (1.0f / HH);
        float dist = sqrtf(rI * rI + rJ * rJ);
        float ang  = (atan2f(rJ, rI) * INV_PI + 1.0f) * 0.5f;
        sa += pw[n] * dist; sb += pw[n] * ang;
    }
    red[tid] = sa; red2[tid] = sb; __syncthreads();
    for (int off = 128; off > 0; off >>= 1) {
        if (tid < off) { red[tid] += red[tid + off]; red2[tid] += red2[tid + off]; }
        __syncthreads();
    }
    const float A = red[0], Bv = red2[0];

    // v[h] = A*w1[0,h] + Bv*w1[1,h];  U[h] = sum_n pw[n]*relu(pw[n]*v[h])
    for (int h = tid; h < HID1; h += 256) {
        float v = A * w1[h] + Bv * w1[HID1 + h];
        float u = 0.f;
        for (int n = 0; n < SPA; ++n) {
            float z = pw[n] * v;
            u += pw[n] * (z > 0.f ? z : 0.f);
        }
        U[b * HID1 + h] = u;
    }
    if (tid == 0) panchor[b] = pw[max_idx];
}

// ---------------- K2b: pack operands for WMMA -------------------------------
// Uh: (16,512) f16, rows 8..15 zero (removes all predication in the GEMM).
// W2t: (768,512) f16, K-contiguous transpose of w2 (512,768) so B-fragment
// halves are 8 consecutive elements -> b128 loads.
__global__ void k_prep_gemm(const float* __restrict__ U,
                            const float* __restrict__ w2,
                            _Float16* __restrict__ Uh,
                            _Float16* __restrict__ W2t) {
    int i = blockIdx.x * blockDim.x + threadIdx.x;   // 768*512 threads
    if (i < 16 * HID1) {
        int m = i >> 9, k = i & 511;
        Uh[i] = (m < BATCH) ? (_Float16)U[m * HID1 + k] : (_Float16)0.0f;
    }
    if (i < HID * HID1) {
        int n = i >> 9, k = i & 511;
        W2t[i] = (_Float16)w2[(size_t)k * HID + n];
    }
}

// ---------------- K3: WMMA GEMM  R = Uh(16x512) @ W2(512x768) ----------------
// gout[b,o] = leaky(panchor[b] * R[b,o]).  One wave per 16x16 N-tile.
// ISA 7.12.2 16-bit A/B layout: lane(lo,hi): elements 0..7 -> K = kb+8*hi..,
// elements 8..15 -> K = kb+16+8*hi..  (two 8-consecutive-K runs per lane).
__global__ void k_wmma_gemm(const _Float16* __restrict__ Uh,   // (16,512)
                            const _Float16* __restrict__ W2t,  // (768,512)
                            const float* __restrict__ panchor, // (8)
                            float* __restrict__ gout) {        // (8,768)
    const int n0   = blockIdx.x * 16;     // 48 tiles over N=768
    const int lane = threadIdx.x;         // 32 = one wave
    const int hi   = lane >> 4;
    const int lo   = lane & 15;
    const _Float16* arow = Uh  + (size_t)lo * HID1 + 8 * hi;          // A row = lo
    const _Float16* brow = W2t + (size_t)(n0 + lo) * HID1 + 8 * hi;   // B col = n0+lo
    v8f acc = {};
#pragma unroll
    for (int kt = 0; kt < 16; ++kt) {
        const int kb = kt * 32;
        v8h a0 = *(const v8h*)(arow + kb);
        v8h a1 = *(const v8h*)(arow + kb + 16);
        v8h b0 = *(const v8h*)(brow + kb);
        v8h b1 = *(const v8h*)(brow + kb + 16);
        v16h a  = __builtin_shufflevector(a0, a1, 0,1,2,3,4,5,6,7,8,9,10,11,12,13,14,15);
        v16h bm = __builtin_shufflevector(b0, b1, 0,1,2,3,4,5,6,7,8,9,10,11,12,13,14,15);
        acc = __builtin_amdgcn_wmma_f32_16x16x32_f16(
                  false, a, false, bm, (short)0, acc, false, false);
    }
    // D layout: lane<16 -> M = r (VGPR idx), N = lane.  Real rows M<8.
    if (hi == 0) {
#pragma unroll
        for (int r = 0; r < 8; ++r) {
            float z = panchor[r] * acc[r];
            gout[r * HID + n0 + lo] = (z > 0.f) ? z : LEAKY_S * z;
        }
    }
}

// ---------------- K4a: 3x3 binomial conv (zero pad) on count ------------------
__global__ void k_conv(const float* __restrict__ count, float* __restrict__ cnt2) {
    const int b = blockIdx.x, tid = threadIdx.x;
    __shared__ float c0[SPA];
    for (int n = tid; n < SPA; n += 256) c0[n] = count[b * SPA + n];
    __syncthreads();
    for (int n = tid; n < SPA; n += 256) {
        int i = n / HH, j = n % HH;
        float s = 0.f;
        for (int di = -1; di <= 1; ++di)
            for (int dj = -1; dj <= 1; ++dj) {
                int ii = i + di, jj = j + dj;
                if (ii >= 0 && ii < HH && jj >= 0 && jj < HH) {
                    float w = ((di == 0) ? 2.f : 1.f) * ((dj == 0) ? 2.f : 1.f);
                    s += w * c0[ii * HH + jj];
                }
            }
        cnt2[b * SPA + n] = s;
    }
}

// ---------------- K4b: per-batch top-84 of smoothed count -> patch_idx --------
__global__ void k_topk_count(const float* __restrict__ cnt2,
                             int* __restrict__ pidx,
                             float* __restrict__ out_pidx) {
    const int b = blockIdx.x, tid = threadIdx.x;
    __shared__ float sw[SPA];
    __shared__ float rv[256];
    __shared__ int   ri[256];
    for (int n = tid; n < SPA; n += 256) sw[n] = cnt2[b * SPA + n];
    __syncthreads();
    for (int it = 0; it < KTOP; ++it) {
        float bv = -INFINITY; int bi = 0x7fffffff;
        for (int n = tid; n < SPA; n += 256) {
            float v = sw[n];
            if (v > bv || (v == bv && n < bi)) { bv = v; bi = n; }
        }
        rv[tid] = bv; ri[tid] = bi; __syncthreads();
        for (int off = 128; off > 0; off >>= 1) {
            if (tid < off) {
                float ov = rv[tid + off]; int oi = ri[tid + off];
                if (ov > rv[tid] || (ov == rv[tid] && oi < ri[tid])) { rv[tid] = ov; ri[tid] = oi; }
            }
            __syncthreads();
        }
        if (tid == 0) {
            int w = ri[0]; sw[w] = -INFINITY;
            pidx[b * KTOP + it]     = w + 1;
            out_pidx[b * KTOP + it] = (float)(w + 1);
        }
        __syncthreads();
    }
}

// ---------------- K5a: copy hidden_states (+ add GCN row at s==0) -------------
__global__ void k_copy_hidden(const float4* __restrict__ hid,
                              const float4* __restrict__ gout,   // (8,192) float4
                              float4* __restrict__ out) {
    const int TOT = BATCH * NROW * (HID / 4);   // 1,205,760 float4
    int i = blockIdx.x * blockDim.x + threadIdx.x;
    if (i >= TOT) return;
    __builtin_prefetch(hid + i + 4096, 0, 0);   // global_prefetch_b8
    float4 v = hid[i];
    const int per_b = NROW * (HID / 4);         // 150,720
    int rem = i % per_b;
    if (rem < (HID / 4)) {                      // row s == 0
        int b = i / per_b;
        float4 g = gout[b * (HID / 4) + rem];
        v.x += g.x; v.y += g.y; v.z += g.z; v.w += g.w;
    }
    out[i] = v;
}

// ---------------- K5b: gather selected_hidden (patch_idx >= 1, unmodified rows)
__global__ void k_gather(const float4* __restrict__ hid,
                         const int* __restrict__ pidx,
                         float4* __restrict__ out_sel) {
    const int bi = blockIdx.x;                  // b*84 + i
    const int b  = bi / KTOP;
    const int r  = pidx[bi];                    // in [1,784]
    const int t  = threadIdx.x;                 // 192 = 768/4
    out_sel[(size_t)bi * (HID / 4) + t] = hid[((size_t)b * NROW + r) * (HID / 4) + t];
}

extern "C" void kernel_launch(void* const* d_in, const int* in_sizes, int n_in,
                              void* d_out, int out_size, void* d_ws, size_t ws_size,
                              hipStream_t stream) {
    (void)in_sizes; (void)n_in; (void)out_size; (void)ws_size;
    const float* hid = (const float*)d_in[0];   // (8,785,768)
    const float* x   = (const float*)d_in[1];   // (8,12,785,785)
    const float* w1  = (const float*)d_in[3];   // (2,512)
    const float* w2  = (const float*)d_in[4];   // (512,768)
    float* out = (float*)d_out;
    float* ws  = (float*)d_ws;

    // workspace layout (floats, all 16B-aligned offsets)
    float*    ns      = ws;                      // 8*12*784 = 75,264
    float*    count   = ws + 75264;              // 8*784    =  6,272
    float*    cnt2    = ws + 81536;              // 8*784
    float*    U       = ws + 87808;              // 8*512
    float*    panchor = ws + 91904;              // 8 (+pad)
    float*    gout    = ws + 91912;              // 8*768
    int*      pidx    = (int*)(ws + 98056);      // 8*84 ints (pad to 98728)
    _Float16* Uh      = (_Float16*)(ws + 98728); // 16*512 halves  (4,096 floats)
    _Float16* W2t     = (_Float16*)(ws + 102824);// 768*512 halves (196,608 floats)
    // ws end: 299,432 floats (~1.2 MB)

    const int HID_FLOATS  = BATCH * NROW * HID;        // 4,823,040
    const int SEL_FLOATS  = BATCH * KTOP * HID;        //   516,096
    float* out_hidden = out;
    float* out_sel    = out + HID_FLOATS;
    float* out_pidx   = out + HID_FLOATS + SEL_FLOATS;

    k_zero        <<<(BATCH * SPA + 255) / 256, 256, 0, stream>>>(count, BATCH * SPA);
    k_score_topk  <<<BATCH * CHAN, 256, 0, stream>>>(x, ns, count);
    k_batch_reduce<<<BATCH, 256, 0, stream>>>(ns, w1, U, panchor);
    k_prep_gemm   <<<(HID * HID1 + 255) / 256, 256, 0, stream>>>(U, w2, Uh, W2t);
    k_wmma_gemm   <<<HID / 16, 32, 0, stream>>>(Uh, W2t, panchor, gout);
    k_conv        <<<BATCH, 256, 0, stream>>>(count, cnt2);
    k_topk_count  <<<BATCH, 256, 0, stream>>>(cnt2, pidx, out_pidx);
    k_copy_hidden <<<(BATCH * NROW * (HID / 4) + 255) / 256, 256, 0, stream>>>(
        (const float4*)hid, (const float4*)gout, (float4*)out_hidden);
    k_gather      <<<BATCH * KTOP, HID / 4, 0, stream>>>(
        (const float4*)hid, pidx, (float4*)out_sel);
}
